// GptOssSparseMoeBlock_45183055953926
// MI455X (gfx1250) — compile-verified, matrix operand order checked by reading, not run on hardware
//
#include <hip/hip_runtime.h>
#include <hip/hip_bf16.h>

// ---------------------------------------------------------------------------
// GPT-OSS sparse MoE block for MI455X (gfx1250, wave32, WMMA bf16 16x16x32).
// Weight-stationary expert GEMMs: each block parks a K=2048 x 32-col bf16
// weight slab (128.5 KB) in LDS (320 KB/WGP) so w1/w2 stream from HBM exactly
// once (~0.8 GB -> ~35 us floor @ 23.3 TB/s). x (pre-converted to bf16), act
// and out live in the 192 MB L2. Each wave computes a 32x32 output tile:
// 2 A frags x 2 B frags -> 4 v_wmma_f32_16x16x32_bf16 per k-step.
// ---------------------------------------------------------------------------

#define E_      16
#define TOPK_   4
#define H_      2048
#define I_      2048
#define ALPHA_  1.702f
#define LIMIT_  7.0f
#define PITCH_  2056            // LDS row pitch (bf16 elems): 16B-aligned rows

typedef __attribute__((ext_vector_type(16))) __bf16 v16bf;
typedef __attribute__((ext_vector_type(8)))  float  v8f;

union FragBF { v16bf v; unsigned short u[16]; uint4 q[2]; };
union AccF   { v8f   v; float f[8]; };

__device__ __forceinline__ unsigned int f2bf2(float lo, float hi) {
#if defined(__has_builtin) && __has_builtin(__builtin_amdgcn_cvt_pk_bf16_f32)
  auto p = __builtin_amdgcn_cvt_pk_bf16_f32(lo, hi);   // v_cvt_pk_bf16_f32
  return __builtin_bit_cast(unsigned int, p);
#else
  unsigned int a = __float_as_uint(lo), b = __float_as_uint(hi);
  a += 0x7FFFu + ((a >> 16) & 1u);                     // RNE
  b += 0x7FFFu + ((b >> 16) & 1u);
  return (a >> 16) | (b & 0xFFFF0000u);
#endif
}

__device__ __forceinline__ unsigned short f2bf(float f) {
  unsigned int b = __float_as_uint(f);
  b += 0x7FFFu + ((b >> 16) & 1u);
  return (unsigned short)(b >> 16);
}

// ---------------------------------------------------------------------------
__global__ void moe_zero_cnt(int* cnt) {
  if (threadIdx.x < E_) cnt[threadIdx.x] = 0;
}

// x (T*H f32) -> xbf (T*H bf16), once per launch; 4 elements per thread.
__global__ void moe_x2bf(const float* __restrict__ x, unsigned int* __restrict__ xbf2) {
  int idx = blockIdx.x * 256 + threadIdx.x;            // index of float4
  float4 p = ((const float4*)x)[idx >> 1];
  // each thread owns one u32 (2 bf16); two threads share one float4 load is
  // wasteful -> process 4 floats per thread instead:
  (void)p;
  size_t i4 = (size_t)idx;                             // index of float4
  float4 q = ((const float4*)x)[i4];
  xbf2[2 * i4 + 0] = f2bf2(q.x, q.y);
  xbf2[2 * i4 + 1] = f2bf2(q.z, q.w);
}

// One block per token: logits over 16 experts, top-4, softmax, routing lists.
__global__ void moe_router(const float* __restrict__ x, const float* __restrict__ rw,
                           const float* __restrict__ rb, float* __restrict__ cw,
                           int* __restrict__ cnt, int* __restrict__ list,
                           float* __restrict__ pw, int T) {
  __shared__ float xs[H_];
  __shared__ float red[E_][16];
  __shared__ float logits[E_];
  const int t = blockIdx.x;
  const int tid = threadIdx.x;
  for (int k = tid; k < H_; k += 256) xs[k] = x[(size_t)t * H_ + k];
  __syncthreads();
  const int e = tid >> 4, j = tid & 15;
  float acc = 0.f;
  const float* rwe = rw + (size_t)e * H_;
  for (int k = j; k < H_; k += 16) acc += xs[k] * rwe[k];
  red[e][j] = acc;
  __syncthreads();
  if (j == 0) {
    float s = 0.f;
    for (int i = 0; i < 16; ++i) s += red[e][i];
    logits[e] = s + rb[e];
  }
  __syncthreads();
  if (tid == 0) {
    float l[E_];
    for (int i = 0; i < E_; ++i) l[i] = logits[i];
    int sel[TOPK_]; float v[TOPK_];
    for (int r = 0; r < TOPK_; ++r) {                  // top-4, first index wins ties
      int bi = 0; float bv = l[0];
      for (int i = 1; i < E_; ++i) if (l[i] > bv) { bv = l[i]; bi = i; }
      sel[r] = bi; v[r] = bv; l[bi] = -1e30f;
    }
    float s = 0.f, w[TOPK_];
    for (int r = 0; r < TOPK_; ++r) { w[r] = __expf(v[r] - v[0]); s += w[r]; }
    float rs = __builtin_amdgcn_rcpf(s);
    float cwrow[E_];
    for (int i = 0; i < E_; ++i) cwrow[i] = 0.f;
    for (int r = 0; r < TOPK_; ++r) {
      float wr = w[r] * rs;
      cwrow[sel[r]] = wr;
      int pos = atomicAdd(&cnt[sel[r]], 1);
      list[sel[r] * T + pos] = t;
      pw[sel[r] * T + pos] = wr;
    }
    for (int i = 0; i < E_; ++i) cw[(size_t)t * E_ + i] = cwrow[i];
  }
}

// 16-wide exclusive scan of expert counts -> compact activation row bases.
__global__ void moe_scan(const int* __restrict__ cnt, int* __restrict__ base) {
  if (threadIdx.x == 0) {
    int s = 0;
    for (int e = 0; e < E_; ++e) { base[e] = s; s += cnt[e]; }
  }
}

// out[t][h] = sum_e cw[t,e] * b2[e][h]  (bias term; fully overwrites d_out)
__global__ void moe_init_out(const float* __restrict__ cw, const float* __restrict__ b2,
                             float* __restrict__ out, int T) {
  int idx = blockIdx.x * 256 + threadIdx.x;
  if (idx >= T * H_) return;
  int t = idx >> 11, h = idx & (H_ - 1);
  float acc = 0.f;
  for (int e = 0; e < E_; ++e) acc += cw[(size_t)t * E_ + e] * b2[(size_t)e * H_ + h];
  out[idx] = acc;
}

// GEMM1 + activation: block = (expert e, 16 gate cols n0). LDS holds the full
// K=2048 slab of 16 gate + 16 up columns (bf16). 8 waves sweep 256-token
// supertiles; each wave: 32 tokens x (16 gate + 16 up), 4 WMMAs / k-step.
__global__ void moe_gemm1(const unsigned short* __restrict__ xbf,
                          const float* __restrict__ w1, const float* __restrict__ b1,
                          const int* __restrict__ cnt, const int* __restrict__ base,
                          const int* __restrict__ list,
                          unsigned short* __restrict__ act, int T) {
  extern __shared__ __align__(16) unsigned short slab[];   // [32][PITCH_]
  const int bx = blockIdx.x;
  const int e  = bx >> 7;                                  // 128 n-tiles / expert
  const int n0 = (bx & 127) << 4;
  const int tid = threadIdx.x;

  // Slab load: pairs of K per thread, packed bf16x2 stores into LDS.
  const size_t w1e = (size_t)e * H_ * (2 * I_);
  for (int idx = tid; idx < (H_ / 2) * 32; idx += 256) {
    int c = idx & 31, k = (idx >> 5) << 1;
    int col = (c < 16) ? (n0 + c) : (I_ + n0 + (c - 16));
    float lo = w1[w1e + (size_t)k * (2 * I_) + col];
    float hi = w1[w1e + (size_t)(k + 1) * (2 * I_) + col];
    *(unsigned int*)(slab + c * PITCH_ + k) = f2bf2(lo, hi);
  }
  __syncthreads();

  const int lane = tid & 31, wave = tid >> 5;
  const int hh = lane >> 4;               // lane half -> K offset group
  const int ml = lane & 15;               // A row within subtile == B column
  const int cntE = cnt[e], baseE = base[e], eT = e * T;
  const float bg = b1[(size_t)e * (2 * I_) + n0 + ml];
  const float bu = b1[(size_t)e * (2 * I_) + I_ + n0 + ml];
  const unsigned short* bgp = slab + ml * PITCH_;
  const unsigned short* bup = slab + (16 + ml) * PITCH_;

  const int mIters = (cntE + 255) >> 8;
  for (int mo = 0; mo < mIters; ++mo) {
    int s0 = mo * 256 + wave * 32 + ml;
    int s1 = s0 + 16;
    int tok0 = (s0 < cntE) ? list[eT + s0] : 0;       // clamped: result unused
    int tok1 = (s1 < cntE) ? list[eT + s1] : 0;
    const unsigned short* xr0 = xbf + (size_t)tok0 * H_;
    const unsigned short* xr1 = xbf + (size_t)tok1 * H_;
    AccF ag0, au0, ag1, au1;
    for (int i = 0; i < 8; ++i) { ag0.f[i]=0.f; au0.f[i]=0.f; ag1.f[i]=0.f; au1.f[i]=0.f; }

#pragma unroll 2
    for (int k0 = 0; k0 < H_; k0 += 32) {
      FragBF a0, a1, bgf, buf_;
      const unsigned short* g = bgp + k0 + hh * 8;
      bgf.q[0] = *(const uint4*)(g);
      bgf.q[1] = *(const uint4*)(g + 16);
      const unsigned short* up = bup + k0 + hh * 8;
      buf_.q[0] = *(const uint4*)(up);
      buf_.q[1] = *(const uint4*)(up + 16);
      const unsigned short* p0 = xr0 + k0 + hh * 8;
      a0.q[0] = *(const uint4*)(p0);
      a0.q[1] = *(const uint4*)(p0 + 16);
      const unsigned short* p1 = xr1 + k0 + hh * 8;
      a1.q[0] = *(const uint4*)(p1);
      a1.q[1] = *(const uint4*)(p1 + 16);

      ag0.v = __builtin_amdgcn_wmma_f32_16x16x32_bf16(false, a0.v, false, bgf.v,
                                                      (short)0, ag0.v, false, false);
      au0.v = __builtin_amdgcn_wmma_f32_16x16x32_bf16(false, a0.v, false, buf_.v,
                                                      (short)0, au0.v, false, false);
      ag1.v = __builtin_amdgcn_wmma_f32_16x16x32_bf16(false, a1.v, false, bgf.v,
                                                      (short)0, ag1.v, false, false);
      au1.v = __builtin_amdgcn_wmma_f32_16x16x32_bf16(false, a1.v, false, buf_.v,
                                                      (short)0, au1.v, false, false);
    }
    // C/D layout: M = r + 8*half, N = lane&15
    for (int sub = 0; sub < 2; ++sub) {
      AccF* agp = sub ? &ag1 : &ag0;
      AccF* aup = sub ? &au1 : &au0;
      for (int r = 0; r < 8; ++r) {
        int sO = mo * 256 + wave * 32 + sub * 16 + r + 8 * hh;
        if (sO < cntE) {
          float g = agp->f[r] + bg;
          float u = aup->f[r] + bu;
          g = fminf(g, LIMIT_);
          u = fminf(fmaxf(u, -LIMIT_), LIMIT_);
          float sg = __builtin_amdgcn_rcpf(1.f + __expf(-ALPHA_ * g));  // v_rcp_f32
          float av = (u + 1.f) * g * sg;
          act[(size_t)(baseE + sO) * I_ + (n0 + ml)] = f2bf(av);
        }
      }
    }
  }
}

// GEMM2 + weighted scatter: block = (expert e, 32 out cols n0). w2 slab in
// LDS, act rows (bf16) stream from L2. Wave tile 32x32 -> 4 WMMAs / k-step.
// out += pw * (act @ w2) via hardware f32 atomics.
__global__ void moe_gemm2(const unsigned short* __restrict__ act,
                          const float* __restrict__ w2,
                          const int* __restrict__ cnt, const int* __restrict__ base,
                          const int* __restrict__ list, const float* __restrict__ pw,
                          float* __restrict__ out, int T) {
  extern __shared__ __align__(16) unsigned short slab[];   // [32][PITCH_]
  const int bx = blockIdx.x;
  const int e  = bx >> 6;                                  // 64 n-tiles / expert
  const int n0 = (bx & 63) << 5;
  const int tid = threadIdx.x;

  const size_t w2e = (size_t)e * I_ * H_;
  for (int idx = tid; idx < (I_ / 2) * 32; idx += 256) {
    int c = idx & 31, k = (idx >> 5) << 1;
    float lo = w2[w2e + (size_t)k * H_ + n0 + c];
    float hi = w2[w2e + (size_t)(k + 1) * H_ + n0 + c];
    *(unsigned int*)(slab + c * PITCH_ + k) = f2bf2(lo, hi);
  }
  __syncthreads();

  const int lane = tid & 31, wave = tid >> 5;
  const int hh = lane >> 4, ml = lane & 15;
  const int cntE = cnt[e], baseE = base[e], eT = e * T;
  const unsigned short* b0p = slab + ml * PITCH_;
  const unsigned short* b1p = slab + (16 + ml) * PITCH_;

  const int mIters = (cntE + 255) >> 8;
  for (int mo = 0; mo < mIters; ++mo) {
    int s0 = mo * 256 + wave * 32 + ml;
    int s1 = s0 + 16;
    int row0 = baseE + ((s0 < cntE) ? s0 : 0);
    int row1 = baseE + ((s1 < cntE) ? s1 : 0);
    const unsigned short* ar0 = act + (size_t)row0 * I_;
    const unsigned short* ar1 = act + (size_t)row1 * I_;
    AccF c00, c01, c10, c11;
    for (int i = 0; i < 8; ++i) { c00.f[i]=0.f; c01.f[i]=0.f; c10.f[i]=0.f; c11.f[i]=0.f; }

#pragma unroll 2
    for (int k0 = 0; k0 < I_; k0 += 32) {
      FragBF a0, a1, bA, bB;
      const unsigned short* g0 = b0p + k0 + hh * 8;
      bA.q[0] = *(const uint4*)(g0);
      bA.q[1] = *(const uint4*)(g0 + 16);
      const unsigned short* g1 = b1p + k0 + hh * 8;
      bB.q[0] = *(const uint4*)(g1);
      bB.q[1] = *(const uint4*)(g1 + 16);
      const unsigned short* p0 = ar0 + k0 + hh * 8;
      a0.q[0] = *(const uint4*)(p0);
      a0.q[1] = *(const uint4*)(p0 + 16);
      const unsigned short* p1 = ar1 + k0 + hh * 8;
      a1.q[0] = *(const uint4*)(p1);
      a1.q[1] = *(const uint4*)(p1 + 16);

      c00.v = __builtin_amdgcn_wmma_f32_16x16x32_bf16(false, a0.v, false, bA.v,
                                                      (short)0, c00.v, false, false);
      c01.v = __builtin_amdgcn_wmma_f32_16x16x32_bf16(false, a0.v, false, bB.v,
                                                      (short)0, c01.v, false, false);
      c10.v = __builtin_amdgcn_wmma_f32_16x16x32_bf16(false, a1.v, false, bA.v,
                                                      (short)0, c10.v, false, false);
      c11.v = __builtin_amdgcn_wmma_f32_16x16x32_bf16(false, a1.v, false, bB.v,
                                                      (short)0, c11.v, false, false);
    }
    for (int sub = 0; sub < 2; ++sub) {
      AccF* cA = sub ? &c10 : &c00;
      AccF* cB = sub ? &c11 : &c01;
      for (int r = 0; r < 8; ++r) {
        int sO = mo * 256 + wave * 32 + sub * 16 + r + 8 * hh;
        if (sO < cntE) {
          int tok = list[eT + sO];
          float wgt = pw[eT + sO];
          float* orow = out + (size_t)tok * H_ + n0;
          unsafeAtomicAdd(orow + ml,      wgt * cA->f[r]);  // global_atomic_add_f32
          unsafeAtomicAdd(orow + 16 + ml, wgt * cB->f[r]);
        }
      }
    }
  }
}

// ---------------------------------------------------------------------------
extern "C" void kernel_launch(void* const* d_in, const int* in_sizes, int n_in,
                              void* d_out, int out_size, void* d_ws, size_t ws_size,
                              hipStream_t stream) {
  const float* x  = (const float*)d_in[0];
  const float* rw = (const float*)d_in[1];
  const float* rb = (const float*)d_in[2];
  const float* w1 = (const float*)d_in[3];
  const float* b1 = (const float*)d_in[4];
  const float* w2 = (const float*)d_in[5];
  const float* b2 = (const float*)d_in[6];
  float* out = (float*)d_out;
  const int T = in_sizes[0] / H_;          // 4096

  // Workspace (~85 MB): counters, bases, routing lists, cw, bf16 act, bf16 x.
  char* ws = (char*)d_ws;
  int*   cnt  = (int*)ws;                          // E (padded to 32)
  int*   base = cnt + 32;                          // E (padded to 32)
  int*   list = base + 32;                         // E*T
  float* pw   = (float*)(list + E_ * T);           // E*T
  float* cw   = pw + (size_t)E_ * T;               // T*E
  unsigned short* act = (unsigned short*)(cw + (size_t)T * E_);   // (4T)*I bf16
  unsigned short* xbf = act + (size_t)(TOPK_ * T) * I_;           // T*H bf16

  const size_t slabBytes = (size_t)32 * PITCH_ * sizeof(unsigned short); // 131584 B
  hipFuncSetAttribute(reinterpret_cast<const void*>(moe_gemm1),
                      hipFuncAttributeMaxDynamicSharedMemorySize, (int)slabBytes);
  hipFuncSetAttribute(reinterpret_cast<const void*>(moe_gemm2),
                      hipFuncAttributeMaxDynamicSharedMemorySize, (int)slabBytes);

  moe_zero_cnt<<<1, 32, 0, stream>>>(cnt);
  moe_x2bf    <<<(T * H_) / (256 * 4), 256, 0, stream>>>(x, (unsigned int*)xbf);
  moe_router  <<<T, 256, 0, stream>>>(x, rw, rb, cw, cnt, list, pw, T);
  moe_scan    <<<1, 32, 0, stream>>>(cnt, base);
  moe_init_out<<<(T * H_ + 255) / 256, 256, 0, stream>>>(cw, b2, out, T);
  moe_gemm1   <<<E_ * (I_ / 16), 256, slabBytes, stream>>>(xbf, w1, b1, cnt, base, list, act, T);
  moe_gemm2   <<<E_ * (H_ / 32), 256, slabBytes, stream>>>(act, w2, cnt, base, list, pw, out, T);
}
